// MultiHeadedAttention_31224412241974
// MI455X (gfx1250) — matrix operand order repaired and checked by reading, not compiled, not run on hardware
//
#include <hip/hip_runtime.h>

// MI455X / gfx1250, wave32. WMMA f32_16x16x32_f16 for all large GEMMs,
// async global->LDS staging (ASYNCcnt) for the weight tiles.
// Roofline: dominant cost is streaming 268MB of v_alpha/v_gamma output
// (~11.5us at 23.3 TB/s); alpha/gamma math is collapsed to 4 FMAs/elem via
// U[b,h,j,:] = v[b,h,j,:] @ W_h precomputation.

typedef __attribute__((ext_vector_type(16))) _Float16 v16h;
typedef __attribute__((ext_vector_type(8)))  float    v8f;

#if __has_builtin(__builtin_amdgcn_global_load_async_to_lds_b128) && \
    __has_builtin(__builtin_amdgcn_s_wait_asynccnt)
#define USE_ASYNC_LDS 1
typedef int v4i_ __attribute__((__vector_size__(4 * sizeof(int))));
typedef __attribute__((address_space(1))) v4i_ as1_v4i;
typedef __attribute__((address_space(3))) v4i_ as3_v4i;
#endif

#define B_    2
#define T_    512
#define D_    512
#define H_    8
#define DK_   64
#define HALF_ 4

__device__ __forceinline__ float softplusf_(float x) {
    return (x > 20.f) ? x : log1pf(__expf(x));
}
__device__ __forceinline__ float sigmoidf_(float x) {
    return 1.f / (1.f + __expf(-x));
}

// ---------------------------------------------------------------------------
// Kernel 0: convert Wq/Wk/Wv f32 -> f16 (one pass, 3 MB read / 1.5 MB write)
// so the projection GEMM can stage f16 weight tiles straight to LDS.
// ---------------------------------------------------------------------------
__global__ __launch_bounds__(256) void wcvt_kernel(
    const float* __restrict__ Wq, const float* __restrict__ Wk,
    const float* __restrict__ Wv, _Float16* __restrict__ Wh)
{
    const int tid = blockIdx.x * 256 + threadIdx.x;   // 0..196607
    const int mat = tid >> 16;                        // 65536 threads / matrix
    const int off = (tid & 65535) * 4;
    const float* W = (mat == 0) ? Wq : (mat == 1) ? Wk : Wv;
    const float4 v = *(const float4*)(W + off);
    _Float16* o = Wh + (size_t)mat * (D_ * D_) + off;
    o[0] = (_Float16)v.x; o[1] = (_Float16)v.y;
    o[2] = (_Float16)v.z; o[3] = (_Float16)v.w;
}

// ---------------------------------------------------------------------------
// Kernel 1: fused Q/K/V projections.  C = X @ W + b, scatter to (b,h,t,dk)
// layout in f16 (WMMA feed) and f32 (V only, needed by U kernel).
// Block (256 thr, 8 waves) owns a 16x128 output tile: A tile (16x32) shared
// by all waves via LDS; W tile (32x128 f16, 8KB) staged via async->LDS b128.
// grid: 3 mats * 64 row-tiles * 4 col-blocks = 768 blocks.
// ---------------------------------------------------------------------------
__global__ __launch_bounds__(256) void proj_kernel(
    const float* __restrict__ query, const float* __restrict__ key_,
    const float* __restrict__ value,
    const _Float16* __restrict__ Wh,
    const float* __restrict__ bq, const float* __restrict__ bk,
    const float* __restrict__ bv,
    _Float16* __restrict__ qf16, _Float16* __restrict__ kf16,
    _Float16* __restrict__ vf16, float* __restrict__ vf32)
{
    __shared__ _Float16 Ah[16 * 40];     // 16x32 A tile, padded rows
    __shared__ _Float16 Wl[32 * 128];    // 32x128 W tile (8 KB)

    const int tid  = threadIdx.x;
    const int lane = tid & 31;
    const int wave = tid >> 5;
    const int mat  = blockIdx.x >> 8;          // 0..2
    const int rem  = blockIdx.x & 255;
    const int tm   = rem >> 2;                 // row-tile 0..63
    const int cb   = rem & 3;                  // col-block (128 cols) 0..3

    const float* X    = (mat == 0) ? query : (mat == 1) ? key_ : value;
    const float* bias = (mat == 0) ? bq : (mat == 1) ? bk : bv;
    const _Float16* Wsrc = Wh + (size_t)mat * (D_ * D_);
    _Float16* O16 = (mat == 0) ? qf16 : (mat == 1) ? kf16 : vf16;

    const int r  = lane & 15;
    const int hl = lane >> 4;

    v8f acc = {};
    for (int k0 = 0; k0 < D_; k0 += 32) {
        // --- stage A tile: 16x32 f32 -> f16, 2 elements per thread ---
#pragma unroll
        for (int t = 0; t < 2; ++t) {
            const int i   = tid + t * 256;     // 0..511
            const int row = i >> 5, kk = i & 31;
            Ah[row * 40 + kk] = (_Float16)X[(tm * 16 + row) * D_ + k0 + kk];
        }
        // --- stage W tile: 32x128 f16 as 512 x 16B chunks, 2 per thread ---
#pragma unroll
        for (int t = 0; t < 2; ++t) {
            const int ci  = tid + t * 256;     // 0..511
            const int row = ci >> 4, c8 = ci & 15;
            const _Float16* gp = Wsrc + (k0 + row) * D_ + cb * 128 + c8 * 8;
            _Float16*       lp = Wl + row * 128 + c8 * 8;
#ifdef USE_ASYNC_LDS
            __builtin_amdgcn_global_load_async_to_lds_b128(
                (as1_v4i*)gp, (as3_v4i*)lp, 0, 0);
#else
            *(uint4*)lp = *(const uint4*)gp;
#endif
        }
#ifdef USE_ASYNC_LDS
        __builtin_amdgcn_s_wait_asynccnt(0);
#endif
        __syncthreads();

        // --- WMMA: each wave owns a 16-col tile of the 128-col block ---
        v16h a, bf;
#pragma unroll
        for (int e = 0; e < 8; ++e) {
            a[e]     = Ah[r * 40 + 8 * hl + e];
            a[8 + e] = Ah[r * 40 + 16 + 8 * hl + e];
        }
#pragma unroll
        for (int e = 0; e < 16; ++e)
            bf[e] = Wl[(8 * hl + e + (e & 8)) * 128 + wave * 16 + r];
        acc = __builtin_amdgcn_wmma_f32_16x16x32_f16(
            false, a, false, bf, (short)0, acc, false, false);
        __syncthreads();
    }

    // --- epilogue: bias + scatter to (b,h,t,dk) ---
    const int n   = lane & 15;
    const int col = cb * 128 + wave * 16 + n;
    const int hh  = col >> 6, dk = col & 63;
    const float bcst = bias[col];
#pragma unroll
    for (int e = 0; e < 8; ++e) {
        const int m = e + 8 * hl;
        const int g = tm * 16 + m;           // global row = b*T + t
        const int bb = g >> 9, tt = g & 511;
        const float v = acc[e] + bcst;
        const int idx = ((bb * H_ + hh) * T_ + tt) * DK_ + dk;
        O16[idx] = (_Float16)v;
        if (mat == 2) vf32[idx] = v;
    }
}

// ---------------------------------------------------------------------------
// Kernel 2: per (b,h, 16-row i-tile): scores (WMMA) -> causal softmax (LDS)
// -> p to global (alpha/gamma needs it) -> v_mu = p @ v (WMMA).
// grid: 512 blocks x 256 thr (8 waves). LDS tile 16 x (512+8) f32 = 33 KB.
// ---------------------------------------------------------------------------
__global__ __launch_bounds__(256) void attn_kernel(
    const _Float16* __restrict__ qf16, const _Float16* __restrict__ kf16,
    const _Float16* __restrict__ vf16,
    float* __restrict__ p32, float* __restrict__ vmu)
{
    __shared__ float S[16][T_ + 8];
    const int lane = threadIdx.x & 31;
    const int wave = threadIdx.x >> 5;
    const int it = blockIdx.x & 31;           // i-tile 0..31
    const int bh = blockIdx.x >> 5;           // 0..15
    const _Float16* qb = qf16 + bh * T_ * DK_;
    const _Float16* kb = kf16 + bh * T_ * DK_;
    const _Float16* vb = vf16 + bh * T_ * DK_;

    const int r  = lane & 15;
    const int hl = lane >> 4;

    // --- Phase A: S = (q @ k^T) * 1/sqrt(DK), causal mask ---
    for (int s = 0; s < 4; ++s) {
        const int jt = wave * 4 + s;          // j-tile 0..31
        v8f acc = {};
#pragma unroll
        for (int k0 = 0; k0 < DK_; k0 += 32) {
            v16h a, bf;
            const _Float16* ap = qb + (it * 16 + r) * DK_ + k0;
#pragma unroll
            for (int e = 0; e < 8; ++e) {
                a[e] = ap[8 * hl + e]; a[8 + e] = ap[16 + 8 * hl + e];
            }
            const _Float16* bp = kb + (jt * 16 + r) * DK_ + k0 + 8 * hl;
#pragma unroll
            for (int e = 0; e < 16; ++e) bf[e] = bp[e + (e & 8)];
            acc = __builtin_amdgcn_wmma_f32_16x16x32_f16(
                false, a, false, bf, (short)0, acc, false, false);
        }
#pragma unroll
        for (int e = 0; e < 8; ++e) {
            const int m  = e + 8 * hl;
            const int ig = it * 16 + m;
            const int jg = jt * 16 + r;
            float v = acc[e] * 0.125f;
            if (jg > ig) v = -1.0e9f;
            S[m][jt * 16 + r] = v;
        }
    }
    __syncthreads();

    // --- Phase B: row softmax (one wave per 2 rows, shuffle reductions) ---
    for (int rr = 0; rr < 2; ++rr) {
        const int m  = wave * 2 + rr;
        const int ig = it * 16 + m;
        float mx = -3.0e38f;
        for (int j = lane; j < T_; j += 32) mx = fmaxf(mx, S[m][j]);
#pragma unroll
        for (int o = 16; o > 0; o >>= 1) mx = fmaxf(mx, __shfl_xor(mx, o, 32));
        float sum = 0.f;
        for (int j = lane; j < T_; j += 32) sum += __expf(S[m][j] - mx);
#pragma unroll
        for (int o = 16; o > 0; o >>= 1) sum += __shfl_xor(sum, o, 32);
        const float inv = 1.f / sum;
        for (int j = lane; j < T_; j += 32) {
            const float p = (j <= ig) ? __expf(S[m][j] - mx) * inv : 0.f;
            S[m][j] = p;
            p32[(bh * T_ + ig) * T_ + j] = p;
        }
    }
    __syncthreads();

    // --- Phase C: vmu = p @ v. 4 n-tiles of 16 -> waves 0..3 (uniform). ---
    if (wave < 4) {
        const int nt = wave;
        v8f acc = {};
        for (int k0 = 0; k0 < T_; k0 += 32) {
            v16h a, bf;
#pragma unroll
            for (int e = 0; e < 8; ++e) {
                a[e]     = (_Float16)S[r][k0 + 8 * hl + e];
                a[8 + e] = (_Float16)S[r][k0 + 16 + 8 * hl + e];
            }
#pragma unroll
            for (int e = 0; e < 16; ++e)
                bf[e] = vb[(k0 + 8 * hl + e + (e & 8)) * DK_ + nt * 16 + r];
            acc = __builtin_amdgcn_wmma_f32_16x16x32_f16(
                false, a, false, bf, (short)0, acc, false, false);
        }
#pragma unroll
        for (int e = 0; e < 8; ++e) {
            const int m = e + 8 * hl;
            vmu[(bh * T_ + it * 16 + m) * DK_ + nt * 16 + r] = acc[e];
        }
    }
}

// ---------------------------------------------------------------------------
// Kernel 3: U[b,h,j,n] = v[b,h(+4),j,:] @ W_h  (tiny, f32 scalar for accuracy)
// 524288 threads; coalesced on n.
// ---------------------------------------------------------------------------
__global__ __launch_bounds__(256) void u_kernel(
    const float* __restrict__ vf32,
    const float* __restrict__ Wa, const float* __restrict__ Wg,
    float* __restrict__ Ua, float* __restrict__ Ug)
{
    const int tid = blockIdx.x * 256 + threadIdx.x;   // 0..524287
    const int n = tid & 63;
    const int j = (tid >> 6) & 511;
    const int h = (tid >> 15) & 3;
    const int b = (tid >> 17) & 1;
    const int mat = tid >> 18;                        // 0 = alpha, 1 = gamma
    const float* W    = mat ? Wg : Wa;
    const float* vrow = vf32 + ((b * H_ + (mat ? h + HALF_ : h)) * T_ + j) * DK_;
    float acc = 0.f;
#pragma unroll 8
    for (int d = 0; d < DK_; ++d)
        acc += vrow[d] * W[(h * DK_ + d) * DK_ + n];
    float* U = mat ? Ug : Ua;
    U[((b * HALF_ + h) * T_ + j) * DK_ + n] = acc;
}

// ---------------------------------------------------------------------------
// Kernel 4 (the bandwidth-bound one): v_alpha / v_gamma outputs.
// alpha[b,i,j,n] = softplus(sum_h p[b,h,i,j]*Ua[b,h,j,n] + ba[n])  (j<=i)
// gamma[b,i,j,n] = softplus(10*(sum_h p[b,h+4,i,j]*Ug[...]+bg[n]))/10
// 64 threads per (b,i,j) pair -> 64-float coalesced stores; zeros above diag.
// ---------------------------------------------------------------------------
__global__ __launch_bounds__(256) void alphagamma_kernel(
    const float* __restrict__ p32,
    const float* __restrict__ Ua, const float* __restrict__ Ug,
    const float* __restrict__ ba, const float* __restrict__ bg,
    float* __restrict__ alphaOut, float* __restrict__ gammaOut)
{
    const int  n    = threadIdx.x & 63;
    const long pair = (long)blockIdx.x * 4 + (threadIdx.x >> 6); // b*262144+i*512+j
    const int  j = (int)(pair & 511);
    const int  i = (int)((pair >> 9) & 511);
    const int  b = (int)(pair >> 18);
    const long obase = pair * 64 + n;

    float av, gv;
    if (j > i) {
        av = 0.f; gv = 0.f;
    } else {
        const float* pb = p32 + (long)b * (H_ * T_ * T_) + (long)i * T_ + j;
        float sa = ba[n], sg = bg[n];
#pragma unroll
        for (int h = 0; h < HALF_; ++h) {
            const float pah = pb[(long)h * (T_ * T_)];
            const float pgh = pb[(long)(h + HALF_) * (T_ * T_)];
            sa += pah * Ua[((b * HALF_ + h) * T_ + j) * DK_ + n];
            sg += pgh * Ug[((b * HALF_ + h) * T_ + j) * DK_ + n];
        }
        av = softplusf_(sa);
        gv = softplusf_(10.f * sg) * 0.1f;
    }
    alphaOut[obase] = av;
    gammaOut[obase] = gv;
}

// ---------------------------------------------------------------------------
// Kernel 5: v_mu_out = sigmoid(mu_in @ Wm + bm); mu_in gathers vmu head-wise.
// ---------------------------------------------------------------------------
__global__ __launch_bounds__(256) void muout_kernel(
    const float* __restrict__ vmu, const float* __restrict__ Wm,
    const float* __restrict__ bm, float* __restrict__ out)
{
    const int tid = blockIdx.x * 256 + threadIdx.x;  // 0..65535
    const int n = tid & 63;
    const int g = tid >> 6;
    const int b = g >> 9, t = g & 511;
    float acc = bm[n];
    for (int c = 0; c < D_; ++c) {
        const int h = c >> 6, d = c & 63;
        acc += vmu[((b * H_ + h) * T_ + t) * DK_ + d] * Wm[c * DK_ + n];
    }
    out[tid] = sigmoidf_(acc);
}

// ---------------------------------------------------------------------------
extern "C" void kernel_launch(void* const* d_in, const int* in_sizes, int n_in,
                              void* d_out, int out_size, void* d_ws, size_t ws_size,
                              hipStream_t stream)
{
    const float* query = (const float*)d_in[0];
    const float* key_  = (const float*)d_in[1];
    const float* value = (const float*)d_in[2];
    // d_in[3] = mask (lower-triangular causal; applied analytically)
    const float* Wq = (const float*)d_in[4];
    const float* bq = (const float*)d_in[5];
    const float* Wk = (const float*)d_in[6];
    const float* bk = (const float*)d_in[7];
    const float* Wv = (const float*)d_in[8];
    const float* bv = (const float*)d_in[9];
    const float* Wa = (const float*)d_in[10];
    const float* ba = (const float*)d_in[11];
    const float* Wg = (const float*)d_in[12];
    const float* bg = (const float*)d_in[13];
    const float* Wm = (const float*)d_in[14];
    const float* bm = (const float*)d_in[15];
    (void)in_sizes; (void)n_in; (void)out_size; (void)ws_size;

    // workspace carve-up (~27 MB total)
    char* ws = (char*)d_ws;
    size_t off = 0;
    auto carve = [&](size_t bytes) -> char* {
        char* p = ws + off;
        off += (bytes + 255) & ~(size_t)255;
        return p;
    };
    const size_t NHD = (size_t)B_ * H_ * T_ * DK_;      // 524288
    _Float16* qf16 = (_Float16*)carve(NHD * 2);
    _Float16* kf16 = (_Float16*)carve(NHD * 2);
    _Float16* vf16 = (_Float16*)carve(NHD * 2);
    _Float16* Wh   = (_Float16*)carve((size_t)3 * D_ * D_ * 2);        // 1.5 MB
    float*    vf32 = (float*)   carve(NHD * 4);
    float*    p32  = (float*)   carve((size_t)B_ * H_ * T_ * T_ * 4);  // 16 MB
    float*    vmu  = (float*)   carve(NHD * 4);
    float*    Ua   = (float*)   carve((size_t)B_ * HALF_ * T_ * DK_ * 4);
    float*    Ug   = (float*)   carve((size_t)B_ * HALF_ * T_ * DK_ * 4);

    float* v_mu_out = (float*)d_out;                        // 65536
    float* v_alpha  = v_mu_out + (size_t)B_ * T_ * DK_;     // +65536
    float* v_gamma  = v_alpha + (size_t)B_ * T_ * T_ * DK_; // +33554432

    wcvt_kernel<<<768, 256, 0, stream>>>(Wq, Wk, Wv, Wh);
    proj_kernel<<<768, 256, 0, stream>>>(query, key_, value, Wh,
                                         bq, bk, bv,
                                         qf16, kf16, vf16, vf32);
    attn_kernel<<<512, 256, 0, stream>>>(qf16, kf16, vf16, p32, vmu);
    u_kernel<<<2048, 256, 0, stream>>>(vf32, Wa, Wg, Ua, Ug);
    alphagamma_kernel<<<131072, 256, 0, stream>>>(p32, Ua, Ug, ba, bg,
                                                  v_alpha, v_gamma);
    muout_kernel<<<256, 256, 0, stream>>>(vmu, Wm, bm, v_mu_out);
}
